// ODConv_3rd_33663953666427
// MI455X (gfx1250) — matrix operand order repaired and checked by reading, not compiled, not run on hardware
//
#include <hip/hip_runtime.h>
#include <hip/hip_bf16.h>
#include <math.h>

#define B_  16
#define C1_ 64
#define C2_ 64
#define H_  160
#define W_  160
#define HID_ 16
#define KQ_ 576           // C1*9, GEMM K dim
#define HW_ (H_*W_)       // 25600
#define EPS_ 1e-5f

typedef __attribute__((ext_vector_type(16))) __bf16 v16bf;
typedef __attribute__((ext_vector_type(8)))  __bf16 v8bf;
typedef __attribute__((ext_vector_type(8)))  float  v8f;

union FragBF16 { v16bf v; v8bf h[2]; };

__device__ __forceinline__ float sigmoidf_(float x) { return 1.0f / (1.0f + __expf(-x)); }

// ---------------------------------------------------------------------------
// Kernel 1: ctx[b][c1] = mean over H*W of x  (1024 blocks, one per (b,c1))
// ---------------------------------------------------------------------------
__global__ __launch_bounds__(256) void ctx_mean_k(const float* __restrict__ x,
                                                  float* __restrict__ ctx) {
  const int bc = blockIdx.x;
  const float* p = x + (long)bc * HW_;
  float s = 0.0f;
  for (int i = threadIdx.x; i < HW_; i += 256) s += p[i];
  for (int o = 16; o > 0; o >>= 1) s += __shfl_down(s, o, 32);
  __shared__ float red[8];
  const int lane = threadIdx.x & 31, wid = threadIdx.x >> 5;
  if (lane == 0) red[wid] = s;
  __syncthreads();
  if (threadIdx.x == 0) {
    float t = 0.0f;
    for (int k = 0; k < 8; ++k) t += red[k];
    ctx[bc] = t * (1.0f / (float)HW_);
  }
}

// ---------------------------------------------------------------------------
// Kernel 2: attention trunk + 4 heads + dynamic bias  (single block, 256 thr)
// ---------------------------------------------------------------------------
__global__ __launch_bounds__(256) void attention_k(
    const float* __restrict__ ctx, const float* __restrict__ Wfc,
    const float* __restrict__ gamma_h, const float* __restrict__ beta_h,
    const float* __restrict__ Wf, const float* __restrict__ bf,
    const float* __restrict__ Ws, const float* __restrict__ bs,
    const float* __restrict__ Wc, const float* __restrict__ bc,
    const float* __restrict__ Ww, const float* __restrict__ bw,
    const float* __restrict__ bk,
    float* __restrict__ attn_f, float* __restrict__ attn_c,
    float* __restrict__ attn_s, float* __restrict__ attn_w,
    float* __restrict__ biasw) {
  __shared__ float s_ctx[B_ * C1_];     // 16x64
  __shared__ float s_h[B_ * HID_];      // 16x16
  __shared__ float s_hb[B_ * HID_];
  __shared__ float s_sc[HID_], s_sh[HID_];
  __shared__ float s_aw[B_ * 4];
  const int tid = threadIdx.x;

  for (int i = tid; i < B_ * C1_; i += 256) s_ctx[i] = ctx[i];
  __syncthreads();

  // h = ctx @ Wfc^T : 16x16, one thread per element
  {
    const int i = tid >> 4, j = tid & 15;
    float acc = 0.0f;
    for (int c = 0; c < C1_; ++c) acc += s_ctx[i * C1_ + c] * Wfc[j * C1_ + c];
    s_h[i * HID_ + j] = acc;
  }
  __syncthreads();

  // batch-norm over batch dim (biased var), per column j
  if (tid < HID_) {
    const int j = tid;
    float mu = 0.0f, m2 = 0.0f;
    for (int i = 0; i < B_; ++i) { float v = s_h[i * HID_ + j]; mu += v; m2 += v * v; }
    mu *= (1.0f / B_); m2 *= (1.0f / B_);
    float var = m2 - mu * mu;
    float sc = gamma_h[j] * rsqrtf(var + EPS_);
    s_sc[j] = sc; s_sh[j] = beta_h[j] - mu * sc;
  }
  __syncthreads();
  {
    const int i = tid >> 4, j = tid & 15;
    float v = s_h[i * HID_ + j] * s_sc[j] + s_sh[j];
    s_hb[i * HID_ + j] = fmaxf(v, 0.0f);
  }
  __syncthreads();

  // attn_f (16x64), attn_c (16x64)
  for (int idx = tid; idx < B_ * C2_; idx += 256) {
    const int i = idx >> 6, c = idx & 63;
    float af = 0.0f, ac = 0.0f;
    for (int j = 0; j < HID_; ++j) {
      float hv = s_hb[i * HID_ + j];
      af += hv * Wf[c * HID_ + j];
      ac += hv * Wc[c * HID_ + j];
    }
    attn_f[idx] = sigmoidf_(af + bf[c]);
    attn_c[idx] = sigmoidf_(ac + bc[c]);
  }
  // attn_s (16x9)
  for (int idx = tid; idx < B_ * 9; idx += 256) {
    const int i = idx / 9, s = idx % 9;
    float a = 0.0f;
    for (int j = 0; j < HID_; ++j) a += s_hb[i * HID_ + j] * Ws[s * HID_ + j];
    attn_s[idx] = sigmoidf_(a + bs[s]);
  }
  // attn_w (16x4) softmax
  if (tid < B_) {
    const int i = tid;
    float l[4], mx = -1e30f;
    for (int k = 0; k < 4; ++k) {
      float a = 0.0f;
      for (int j = 0; j < HID_; ++j) a += s_hb[i * HID_ + j] * Ww[k * HID_ + j];
      l[k] = a + bw[k]; mx = fmaxf(mx, l[k]);
    }
    float sm = 0.0f;
    for (int k = 0; k < 4; ++k) { l[k] = __expf(l[k] - mx); sm += l[k]; }
    float inv = 1.0f / sm;
    for (int k = 0; k < 4; ++k) { l[k] *= inv; s_aw[i * 4 + k] = l[k]; attn_w[i * 4 + k] = l[k]; }
  }
  __syncthreads();
  // biasw[b][c2] = attn_w @ bk
  for (int idx = tid; idx < B_ * C2_; idx += 256) {
    const int b = idx >> 6, c2 = idx & 63;
    float a = 0.0f;
    for (int k = 0; k < 4; ++k) a += s_aw[b * 4 + k] * bk[k * C2_ + c2];
    biasw[idx] = a;
  }
}

// ---------------------------------------------------------------------------
// Kernel 3: build per-sample weights in bf16, layout [b][c2][(kh*3+kw)*64+c1]
// grid = B*C2 blocks
// ---------------------------------------------------------------------------
__global__ __launch_bounds__(256) void wbuild_k(
    const float* __restrict__ Wk, const float* __restrict__ attn_f,
    const float* __restrict__ attn_c, const float* __restrict__ attn_s,
    const float* __restrict__ attn_w, __bf16* __restrict__ wbf) {
  const int b = blockIdx.x >> 6;
  const int c2 = blockIdx.x & 63;
  __shared__ float aw[4], as9[9], af;
  if (threadIdx.x < 4) aw[threadIdx.x] = attn_w[b * 4 + threadIdx.x];
  if (threadIdx.x < 9) as9[threadIdx.x] = attn_s[b * 9 + threadIdx.x];
  if (threadIdx.x == 31) af = attn_f[b * C2_ + c2];
  __syncthreads();
  for (int t = threadIdx.x; t < KQ_; t += 256) {
    const int s = t >> 6, c1 = t & 63;            // k = s*64 + c1
    float sum = 0.0f;
    for (int k = 0; k < 4; ++k)
      sum += aw[k] * Wk[((long)(k * C2_ + c2) * C1_ + c1) * 9 + s];
    float v = sum * af * attn_c[b * C1_ + c1] * as9[s];
    wbf[(long)(b * C2_ + c2) * KQ_ + t] = (__bf16)v;
  }
}

// ---------------------------------------------------------------------------
// Kernel 4: implicit-GEMM dynamic conv via v_wmma_f32_16x16x32_bf16
// grid = (W/32, H/2, B), block = 256 (8 waves = 4 c2-tiles x 2 col-halves)
// Each wave keeps TWO accumulators (2 output pixel rows) reusing each A frag.
// ---------------------------------------------------------------------------
#define XPAD 72   // padded c1 stride in LDS slab (16B-align + bank spread)

__global__ __launch_bounds__(256) void conv_wmma_k(
    const float* __restrict__ x, const __bf16* __restrict__ wbf,
    const float* __restrict__ biasw, float* __restrict__ out_pre,
    float* __restrict__ s1, float* __restrict__ s2) {
  const int b  = blockIdx.z;
  const int h0 = blockIdx.y * 2;        // two output rows per block
  const int w0 = blockIdx.x * 32;
  const int tid  = threadIdx.x;
  const int lane = tid & 31;
  const int wid  = tid >> 5;

  __shared__ __bf16 xs[4 * 34 * XPAD];  // [r = h0-1+r][w'][c1], bf16
  __shared__ float st1[C2_], st2[C2_];
  __shared__ float sbias[C2_];
  if (tid < C2_) {
    st1[tid] = 0.0f; st2[tid] = 0.0f;
    sbias[tid] = biasw[b * C2_ + tid];
  }

  // stage x slab: rows h0-1..h0+2, cols w0-1..w0+32, all 64 c1, as bf16
  for (int row = wid; row < 4 * C1_; row += 8) {
    const int c1 = row & 63, r = row >> 6;        // r in 0..3
    const int h = h0 + r - 1;
    const long base = ((long)(b * C1_ + c1) * H_ + h) * W_;
    {
      const int wv = w0 - 1 + lane;
      float v = 0.0f;
      if (h >= 0 && h < H_ && wv >= 0 && wv < W_) v = x[base + wv];
      xs[(r * 34 + lane) * XPAD + c1] = (__bf16)v;
    }
    if (lane < 2) {
      const int wp = 32 + lane;
      const int wv = w0 - 1 + wp;
      float v = 0.0f;
      if (h >= 0 && h < H_ && wv >= 0 && wv < W_) v = x[base + wv];
      xs[(r * 34 + wp) * XPAD + c1] = (__bf16)v;
    }
  }
  __syncthreads();

  const int mTile = wid >> 1;           // c2 tile 0..3
  const int nBase = (wid & 1) * 16;     // pixel column half 0/16
  const int nloc  = lane & 15;
  const int c2A   = mTile * 16 + nloc;        // A-matrix row (M = lane&15)
  const int selA  = (lane < 16) ? 0 : 16;     // A K-half per lane group
  const int selB  = (lane < 16) ? 0 : 16;     // B K-half per lane group
  const __bf16* aBase = wbf + (long)(b * C2_ + c2A) * KQ_;

  v8f acc0 = {};   // output row h0
  v8f acc1 = {};   // output row h0+1
  #pragma unroll
  for (int s = 0; s < 9; ++s) {
    const int kh = s / 3, kw = s % 3;
    #pragma unroll
    for (int g = 0; g < 2; ++g) {
      const int kc = s * 64 + g * 32;
      FragBF16 fa, fb0, fb1;
      const v8bf* pa = (const v8bf*)(aBase + kc + selA);
      fa.h[0] = pa[0];      // K = kc+sel    .. +7   (elems 0..7)
      fa.h[1] = pa[1];      // K = kc+sel+8  .. +15  (elems 8..15)
      const int wq = nBase + nloc + kw;
      const int cb = g * 32 + selB;
      const v8bf* pb0 = (const v8bf*)(&xs[((kh    ) * 34 + wq) * XPAD + cb]);
      const v8bf* pb1 = (const v8bf*)(&xs[((kh + 1) * 34 + wq) * XPAD + cb]);
      fb0.h[0] = pb0[0]; fb0.h[1] = pb0[1];
      fb1.h[0] = pb1[0]; fb1.h[1] = pb1[1];
      acc0 = __builtin_amdgcn_wmma_f32_16x16x32_bf16(
          false, fa.v, false, fb0.v, (short)0, acc0, false, false);
      acc1 = __builtin_amdgcn_wmma_f32_16x16x32_bf16(
          false, fa.v, false, fb1.v, (short)0, acc1, false, false);
    }
  }

  // epilogue: bias add, store out_pre, per-c2 BN partial stats via
  // cross-lane butterfly reduction (N dim lives across the 16-lane halves)
  const int n = nBase + nloc;
  #pragma unroll
  for (int i = 0; i < 8; ++i) {
    const int c2 = mTile * 16 + ((lane < 16) ? i : i + 8);
    const float bias = sbias[c2];
    float v0 = acc0[i] + bias;
    float v1 = acc1[i] + bias;
    const long obase = (long)(b * C2_ + c2) * HW_ + w0 + n;
    out_pre[obase + (h0    ) * W_] = v0;
    out_pre[obase + (h0 + 1) * W_] = v1;
    float sv = v0 + v1;
    float sq = v0 * v0 + v1 * v1;
    #pragma unroll
    for (int m = 8; m > 0; m >>= 1) {       // reduce within each 16-lane half
      sv += __shfl_xor(sv, m, 32);
      sq += __shfl_xor(sq, m, 32);
    }
    if ((lane & 15) == 0) {                 // lanes 0 and 16 own the two c2 rows
      atomicAdd(&st1[c2], sv);
      atomicAdd(&st2[c2], sq);
    }
  }
  __syncthreads();
  if (tid < C2_) {
#if defined(__HIP_DEVICE_COMPILE__)
    unsafeAtomicAdd(&s1[tid], st1[tid]);
    unsafeAtomicAdd(&s2[tid], st2[tid]);
#endif
  }
}

// ---------------------------------------------------------------------------
// Kernel 5: outer BatchNorm + SiLU, float4 streaming
// ---------------------------------------------------------------------------
__global__ __launch_bounds__(256) void finalize_k(
    const float* __restrict__ out_pre, const float* __restrict__ s1,
    const float* __restrict__ s2, const float* __restrict__ gamma_o,
    const float* __restrict__ beta_o, float* __restrict__ out) {
  __shared__ float sc[C2_], sh[C2_];
  if (threadIdx.x < C2_) {
    const float invN = 1.0f / ((float)B_ * (float)HW_);
    float mu = s1[threadIdx.x] * invN;
    float var = s2[threadIdx.x] * invN - mu * mu;
    float scale = gamma_o[threadIdx.x] * rsqrtf(var + EPS_);
    sc[threadIdx.x] = scale;
    sh[threadIdx.x] = beta_o[threadIdx.x] - mu * scale;
  }
  __syncthreads();
  const long total4 = (long)B_ * C2_ * HW_ / 4;       // 6,553,600
  for (long i = (long)blockIdx.x * 256 + threadIdx.x; i < total4;
       i += (long)gridDim.x * 256) {
    float4 v = ((const float4*)out_pre)[i];
    const int c2 = (int)((i / (HW_ / 4)) & 63);
    const float a = sc[c2], bsh = sh[c2];
    float y;
    y = v.x * a + bsh; v.x = y * sigmoidf_(y);
    y = v.y * a + bsh; v.y = y * sigmoidf_(y);
    y = v.z * a + bsh; v.z = y * sigmoidf_(y);
    y = v.w * a + bsh; v.w = y * sigmoidf_(y);
    ((float4*)out)[i] = v;
  }
}

// ---------------------------------------------------------------------------
extern "C" void kernel_launch(void* const* d_in, const int* in_sizes, int n_in,
                              void* d_out, int out_size, void* d_ws, size_t ws_size,
                              hipStream_t stream) {
  const float* x       = (const float*)d_in[0];
  const float* Wk      = (const float*)d_in[1];
  const float* bk      = (const float*)d_in[2];
  const float* Wfc     = (const float*)d_in[3];
  const float* gamma_h = (const float*)d_in[4];
  const float* beta_h  = (const float*)d_in[5];
  const float* Wf      = (const float*)d_in[6];
  const float* bf      = (const float*)d_in[7];
  const float* Ws      = (const float*)d_in[8];
  const float* bs      = (const float*)d_in[9];
  const float* Wc      = (const float*)d_in[10];
  const float* bc      = (const float*)d_in[11];
  const float* Ww      = (const float*)d_in[12];
  const float* bw      = (const float*)d_in[13];
  const float* gamma_o = (const float*)d_in[14];
  const float* beta_o  = (const float*)d_in[15];

  char* ws = (char*)d_ws;
  float*  ctx    = (float*)(ws + 0);          // 1024 f
  float*  attn_f = (float*)(ws + 4096);       // 1024 f
  float*  attn_c = (float*)(ws + 8192);       // 1024 f
  float*  attn_s = (float*)(ws + 12288);      // 144 f
  float*  attn_w = (float*)(ws + 12864);      // 64 f
  float*  biasw  = (float*)(ws + 13312);      // 1024 f
  float*  s1     = (float*)(ws + 17408);      // 64 f
  float*  s2     = (float*)(ws + 17664);      // 64 f
  __bf16* wbf    = (__bf16*)(ws + 18432);     // 16*64*576 bf16 = 1,179,648 B
  float*  outp   = (float*)(ws + 1198080);    // 26,214,400 f = 104,857,600 B
  const size_t need = 1198080ull + 104857600ull;
  if (ws_size < need) return;

  hipMemsetAsync(s1, 0, 512, stream);  // zero s1+s2 (contiguous)

  ctx_mean_k<<<B_ * C1_, 256, 0, stream>>>(x, ctx);
  attention_k<<<1, 256, 0, stream>>>(ctx, Wfc, gamma_h, beta_h, Wf, bf, Ws, bs,
                                     Wc, bc, Ww, bw, bk,
                                     attn_f, attn_c, attn_s, attn_w, biasw);
  wbuild_k<<<B_ * C2_, 256, 0, stream>>>(Wk, attn_f, attn_c, attn_s, attn_w, wbf);
  dim3 g3(W_ / 32, H_ / 2, B_);
  conv_wmma_k<<<g3, 256, 0, stream>>>(x, wbf, biasw, outp, s1, s2);
  finalize_k<<<4096, 256, 0, stream>>>(outp, s1, s2, gamma_o, beta_o, (float*)d_out);
}
